// SelfAttention_47820165874215
// MI455X (gfx1250) — matrix operand order repaired and checked by reading, not compiled, not run on hardware
//
#include <hip/hip_runtime.h>
#include <hip/hip_bf16.h>
#include <cstddef>

#define DIM   1024
#define SEQ   2048
#define BATCH 4
#define KT    256              // keys per streaming tile in attention

// padded LDS row strides (avoid 16-way bank conflicts on ds_load_b128:
// stride mod 64 dwords == 4 -> 16 consecutive rows hit 16 distinct bank groups)
#define QSTR  1032             // bf16 elems: 2064 B = 516 dwords, 516 % 64 = 4
#define PSTR  264              // bf16 elems:  528 B = 132 dwords, 132 % 64 = 4
#define SSTR  260              // f32  elems: 1040 B = 260 dwords, 260 % 64 = 4

typedef __attribute__((ext_vector_type(16))) __bf16 v16bf;
typedef __attribute__((ext_vector_type(8)))  __bf16 v8bf;
typedef __attribute__((ext_vector_type(8)))  float  v8f;

// Build a 16-element bf16 fragment from two 16-byte chunks.
__device__ __forceinline__ v16bf ld16(const __bf16* p0, const __bf16* p1) {
    v8bf lo = *(const v8bf*)p0;
    v8bf hi = *(const v8bf*)p1;
    return __builtin_shufflevector(lo, hi, 0,1,2,3,4,5,6,7,8,9,10,11,12,13,14,15);
}

__device__ __forceinline__ v8f wmma_bf16(v16bf a, v16bf b, v8f c) {
    // D = A(16x32 bf16) * B(32x16 bf16) + C(16x16 f32)
    return __builtin_amdgcn_wmma_f32_16x16x32_bf16(
        /*neg_a=*/false, a, /*neg_b=*/false, b,
        /*c_mod=*/(short)0, c, /*reuse_a=*/false, /*reuse_b=*/false);
}

// ---------------------------------------------------------------- prep kernels
__global__ void cvt_bf16_kernel(const float* __restrict__ in,
                                __bf16* __restrict__ out) {
    const size_t i = (size_t)blockIdx.x * blockDim.x + threadIdx.x;
    out[i] = (__bf16)in[i];
}

// Wt[n][k] = (bf16) W[k][n]   (W is [D_in][D_out] row-major)
__global__ void transpose_w_kernel(const float* __restrict__ W,
                                   __bf16* __restrict__ Wt) {
    const int id = blockIdx.x * 256 + threadIdx.x;
    const int k  = id & (DIM - 1);
    const int n  = id >> 10;
    Wt[(size_t)n * DIM + k] = (__bf16)W[(size_t)k * DIM + n];
}

// ---------------------------------------------------------------- projection GEMM
// Y[m][n] = sum_k X[m][k] * Wt[n][k] + bias[n]
// block = 256 threads = 8 waves (4 in M x 2 in N); block tile 128x128;
// wave tile 32(M) x 64(N) = 2x4 WMMA accumulators; K-step 32.
template <bool TRANS_OUT>
__global__ void __launch_bounds__(256)
proj_gemm_kernel(const __bf16* __restrict__ X,   // [BATCH*SEQ][DIM] bf16 row-major
                 const __bf16* __restrict__ Wt,  // [DIM][DIM] bf16 (n-major)
                 const float*  __restrict__ bias,
                 __bf16* __restrict__ Y)         // row-major, or Vt[b][d][s] if TRANS_OUT
{
    const int lane = threadIdx.x & 31;
    const int wv   = threadIdx.x >> 5;
    const int wm   = wv >> 1;            // 0..3
    const int wn   = wv & 1;             // 0..1
    const int half = lane >> 4;
    const int l16  = lane & 15;

    const int mW = blockIdx.y * 128 + wm * 32;
    const int nW = blockIdx.x * 128 + wn * 64;

    v8f acc[2][4] = {};

    for (int kb = 0; kb < DIM; kb += 32) {
        v16bf a[2], bf[4];
#pragma unroll
        for (int mt = 0; mt < 2; ++mt) {
            const __bf16* pa = X + (size_t)(mW + mt * 16 + l16) * DIM + kb;
            a[mt] = ld16(pa + 8 * half, pa + 16 + 8 * half);   // A layout (ISA 7.12.2)
        }
#pragma unroll
        for (int nt = 0; nt < 4; ++nt) {
            const __bf16* pb = Wt + (size_t)(nW + nt * 16 + l16) * DIM + kb;
            bf[nt] = ld16(pb + 8 * half, pb + 16 + 8 * half);   // B layout: same k order as A
        }
#pragma unroll
        for (int mt = 0; mt < 2; ++mt)
#pragma unroll
            for (int nt = 0; nt < 4; ++nt)
                acc[mt][nt] = wmma_bf16(a[mt], bf[nt], acc[mt][nt]);
    }

#pragma unroll
    for (int nt = 0; nt < 4; ++nt) {
        const int n = nW + nt * 16 + l16;
        const float bn = bias[n];
#pragma unroll
        for (int mt = 0; mt < 2; ++mt) {
            if (TRANS_OUT) {
                v8bf pack;
#pragma unroll
                for (int r = 0; r < 8; ++r) pack[r] = (__bf16)(acc[mt][nt][r] + bn);
                const int m = mW + mt * 16 + 8 * half;   // 8 consecutive s positions
                const int b = m >> 11;                   // m / SEQ
                const int s = m & (SEQ - 1);
                *(v8bf*)(Y + ((size_t)b * DIM + n) * SEQ + s) = pack;
            } else {
#pragma unroll
                for (int r = 0; r < 8; ++r) {
                    const int m = mW + mt * 16 + 8 * half + r;   // C/D: row = r + 8*half
                    Y[(size_t)m * DIM + n] = (__bf16)(acc[mt][nt][r] + bn);
                }
            }
        }
    }
}

// ---------------------------------------------------------------- flash attention
// block = 512 threads = 16 waves; one (batch, 32-query) tile per block.
// Q tile staged once in LDS (padded rows, conflict-free A-fragment loads).
// Streams 256-key tiles: scores (wave w -> score cols [16w,16w+16)) ->
// parallel online softmax (16 lanes/row, shfl_xor reductions) ->
// P*V (wave w -> 64-wide d-slice, accO = 2x4 tiles = 64 VGPRs).
__global__ void __launch_bounds__(512)
attn_kernel(const __bf16* __restrict__ Qb,   // [BATCH*SEQ][DIM]
            const __bf16* __restrict__ Kb,   // [BATCH*SEQ][DIM]
            const __bf16* __restrict__ Vt,   // [BATCH][DIM][SEQ]
            float* __restrict__ out)         // [BATCH*SEQ][DIM] f32
{
    __shared__ __align__(16) __bf16 Qsh[32 * QSTR];
    __shared__ __align__(16) float  Ssh[32 * SSTR];
    __shared__ __align__(16) __bf16 Psh[32 * PSTR];
    __shared__ float mstat[32], lstat[32], alphash[32];

    const int lane = threadIdx.x & 31;
    const int wv   = threadIdx.x >> 5;    // 0..15
    const int half = lane >> 4;
    const int l16  = lane & 15;

    const int nQT = SEQ / 32;
    const int b   = blockIdx.x / nQT;
    const int q0  = (blockIdx.x % nQT) * 32;
    const size_t bOff = (size_t)b * SEQ;

    // ---- stage the 32x1024 Q tile into LDS (padded rows), coalesced 16B/lane
    {
        const __bf16* src = Qb + (bOff + q0) * DIM;
#pragma unroll
        for (int i = 0; i < 8; ++i) {
            const int c   = threadIdx.x + i * 512;   // 4096 chunks of 8 elems
            const int row = c >> 7;
            const int off = (c & 127) * 8;
            *(v8bf*)(Qsh + row * QSTR + off) = *(const v8bf*)(src + (size_t)row * DIM + off);
        }
    }
    if (threadIdx.x < 32) { mstat[threadIdx.x] = -1e30f; lstat[threadIdx.x] = 0.0f; }
    __syncthreads();

    v8f accO[2][4] = {};
    const float scale = 0.03125f;   // 1/sqrt(1024)
    const int dW = wv * 64;         // this wave's output d-slice

    for (int kt = 0; kt < SEQ; kt += KT) {
        // ---- scores: wave wv computes score columns [wv*16, wv*16+16)
        v8f accS[2] = {};
        const __bf16* kRow = Kb + (bOff + kt + wv * 16 + l16) * DIM;
#pragma unroll 2
        for (int kb = 0; kb < DIM; kb += 32) {
            const __bf16* pa0 = Qsh + l16 * QSTR + kb;          // LDS, conflict-free
            const __bf16* pa1 = pa0 + 16 * QSTR;
            v16bf a0 = ld16(pa0 + 8 * half, pa0 + 16 + 8 * half);
            v16bf a1 = ld16(pa1 + 8 * half, pa1 + 16 + 8 * half);
            const __bf16* pb = kRow + kb;
            v16bf bf = ld16(pb + 8 * half, pb + 16 + 8 * half);
            accS[0] = wmma_bf16(a0, bf, accS[0]);
            accS[1] = wmma_bf16(a1, bf, accS[1]);
        }
#pragma unroll
        for (int mt = 0; mt < 2; ++mt)
#pragma unroll
            for (int r = 0; r < 8; ++r)
                Ssh[(mt * 16 + 8 * half + r) * SSTR + wv * 16 + l16] = accS[mt][r] * scale;
        __syncthreads();

        // ---- parallel online softmax: 16 lanes per row, 2 rows per wave
        {
            const int row = wv * 2 + (lane >> 4);
            const int sub = lane & 15;
            const float* sr = &Ssh[row * SSTR + sub * 16];
            float mx = -1e30f;
#pragma unroll
            for (int j = 0; j < 16; ++j) mx = fmaxf(mx, sr[j]);
#pragma unroll
            for (int o = 8; o >= 1; o >>= 1) mx = fmaxf(mx, __shfl_xor(mx, o, 32));
            const float mo = mstat[row];
            const float mn = fmaxf(mo, mx);
            float ps = 0.0f;
            __bf16* pr = &Psh[row * PSTR + sub * 16];
#pragma unroll
            for (int j = 0; j < 16; ++j) {
                const float p = __expf(sr[j] - mn);
                ps += p;
                pr[j] = (__bf16)p;
            }
#pragma unroll
            for (int o = 8; o >= 1; o >>= 1) ps += __shfl_xor(ps, o, 32);
            if (sub == 0) {
                const float alpha = __expf(mo - mn);
                lstat[row]   = lstat[row] * alpha + ps;
                mstat[row]   = mn;
                alphash[row] = alpha;
            }
        }
        __syncthreads();

        // ---- rescale accumulators, then P (LDS) @ Vt (global)
        float al[16];
#pragma unroll
        for (int i = 0; i < 16; ++i)
            al[i] = alphash[(i >> 3) * 16 + 8 * half + (i & 7)];
#pragma unroll
        for (int mt = 0; mt < 2; ++mt)
#pragma unroll
            for (int nt = 0; nt < 4; ++nt)
#pragma unroll
                for (int r = 0; r < 8; ++r)
                    accO[mt][nt][r] *= al[mt * 8 + r];

        for (int kc = 0; kc < KT; kc += 32) {
            v16bf a[2], bfr[4];
#pragma unroll
            for (int mt = 0; mt < 2; ++mt) {
                const __bf16* pa = &Psh[(mt * 16 + l16) * PSTR + kc];
                a[mt] = ld16(pa + 8 * half, pa + 16 + 8 * half);
            }
#pragma unroll
            for (int nt = 0; nt < 4; ++nt) {
                const int n = dW + nt * 16 + l16;
                const __bf16* pb = Vt + ((size_t)b * DIM + n) * SEQ + kt + kc;
                bfr[nt] = ld16(pb + 8 * half, pb + 16 + 8 * half);
            }
#pragma unroll
            for (int mt = 0; mt < 2; ++mt)
#pragma unroll
                for (int nt = 0; nt < 4; ++nt)
                    accO[mt][nt] = wmma_bf16(a[mt], bfr[nt], accO[mt][nt]);
        }
        __syncthreads();
    }

    // ---- epilogue: divide by softmax denominator, store f32
    float linv[16];
#pragma unroll
    for (int i = 0; i < 16; ++i)
        linv[i] = 1.0f / lstat[(i >> 3) * 16 + 8 * half + (i & 7)];
#pragma unroll
    for (int mt = 0; mt < 2; ++mt)
#pragma unroll
        for (int nt = 0; nt < 4; ++nt)
#pragma unroll
            for (int r = 0; r < 8; ++r) {
                const int m = q0 + mt * 16 + 8 * half + r;
                const int n = dW + nt * 16 + l16;
                out[(bOff + m) * DIM + n] = accO[mt][nt][r] * linv[mt * 8 + r];
            }
}

// ---------------------------------------------------------------- launcher
extern "C" void kernel_launch(void* const* d_in, const int* in_sizes, int n_in,
                              void* d_out, int out_size, void* d_ws, size_t ws_size,
                              hipStream_t stream) {
    const float* x  = (const float*)d_in[0];
    const float* Wq = (const float*)d_in[1];
    const float* bq = (const float*)d_in[2];
    const float* Wk = (const float*)d_in[3];
    const float* bk = (const float*)d_in[4];
    const float* Wv = (const float*)d_in[5];
    const float* bv = (const float*)d_in[6];
    float* out = (float*)d_out;

    const size_t nX = (size_t)BATCH * SEQ * DIM;   // 8M elems
    const size_t nW = (size_t)DIM * DIM;           // 1M elems
    __bf16* xbf = (__bf16*)d_ws;
    __bf16* wqt = xbf + nX;
    __bf16* wkt = wqt + nW;
    __bf16* wvt = wkt + nW;
    __bf16* qb  = wvt + nW;
    __bf16* kb  = qb + nX;
    __bf16* vt  = kb + nX;   // Vt layout [BATCH][DIM][SEQ]
    // total workspace: (4*8M + 3*1M) * 2 B ~= 70 MB

    cvt_bf16_kernel<<<(int)(nX / 256), 256, 0, stream>>>(x, xbf);
    transpose_w_kernel<<<(int)(nW / 256), 256, 0, stream>>>(Wq, wqt);
    transpose_w_kernel<<<(int)(nW / 256), 256, 0, stream>>>(Wk, wkt);
    transpose_w_kernel<<<(int)(nW / 256), 256, 0, stream>>>(Wv, wvt);

    dim3 ggrid(DIM / 128, (BATCH * SEQ) / 128);    // (8, 64)
    proj_gemm_kernel<false><<<ggrid, 256, 0, stream>>>(xbf, wqt, bq, qb);
    proj_gemm_kernel<false><<<ggrid, 256, 0, stream>>>(xbf, wkt, bk, kb);
    proj_gemm_kernel<true ><<<ggrid, 256, 0, stream>>>(xbf, wvt, bv, vt);

    attn_kernel<<<BATCH * (SEQ / 32), 512, 0, stream>>>(qb, kb, vt, out);
}
